// Transformer_17377437680208
// MI455X (gfx1250) — matrix operand order repaired and checked by reading, not compiled, not run on hardware
//
#include <hip/hip_runtime.h>
#include <hip/hip_bf16.h>

// ---------------- model constants ----------------
#define BB   2
#define SS   512
#define DMM  512
#define DKK  64
#define DVV  64
#define HH   8
#define FFF  2048
#define NBB  6
#define VV   32000

#define NEGINF (-__builtin_inff())

typedef __attribute__((ext_vector_type(16))) _Float16 v16h;
typedef __attribute__((ext_vector_type(8)))  _Float16 v8h;
typedef __attribute__((ext_vector_type(4)))  _Float16 v4h;
typedef __attribute__((ext_vector_type(8)))  float    v8f;
typedef long long ll;

#if __has_builtin(__builtin_amdgcn_global_load_async_to_lds_b128)
#define HAVE_ASYNC_LDS 1
// exact parameter types per hipcc diagnostic: v4i32 in addrspace(1)/(3)
typedef int v4i_ __attribute__((vector_size(16)));
typedef __attribute__((address_space(1))) v4i_* as1_v4i;
typedef __attribute__((address_space(3))) v4i_* as3_v4i;
#else
#define HAVE_ASYNC_LDS 0
#endif

__device__ __forceinline__ void async_wait0() {
#if __has_builtin(__builtin_amdgcn_s_wait_asynccnt)
  __builtin_amdgcn_s_wait_asynccnt(0);
#else
  asm volatile("s_wait_asynccnt 0x0" ::: "memory");
#endif
}

// ---------------- generic batched WMMA GEMM ----------------
// C[z] = epilogue( scale * A[z] x B[z] )  with z = blockIdx.z,
// zb = z >> hshift, zh = z & ((1<<hshift)-1); offsets: ptr + zb*sb + zh*sh.
// All M % 64 == 0, N % 64 == 0, K % 32 == 0 in this model.
struct GemmP {
  const float* A;  ll asb, ash; int lda;
  const float* Bm; ll bsb, bsh; int ldb;
  float*       C;  ll csb, csh; int ldc;
  const float* bias; ll biasb, biash;
  const float* res;  ll resb,  resh;
  const float* mrow; ll mrowb;
  const float* mcol; ll mcolb;
  int M, N, K, hshift;
  int transB, causal, relu;
  float scale;
};

#define TBM 64
#define TBN 64
#define TBK 32
#define ALD   40  // f16 stride (80B): 16B-aligned + bank skew (fallback path)
#define ALD32 36  // f32 stride (144B): 16B-aligned + bank skew (async path)
#define BLD   40

template <int TRANSB>
__launch_bounds__(256)
__global__ void gemm_wmma(GemmP p) {
#if HAVE_ASYNC_LDS
  __shared__ __align__(16) float As32[TBM * ALD32];  // [m][k] f32, filled by async DMA
#else
  __shared__ __align__(16) _Float16 As[TBM * ALD];   // [m][k] f16
#endif
  __shared__ __align__(16) _Float16 Bs[TBN * BLD];   // [n][k] f16 (transposed stage)

  const int z  = blockIdx.z;
  const int zb = z >> p.hshift, zh = z & ((1 << p.hshift) - 1);
  const float* __restrict__ A  = p.A  + (ll)zb * p.asb + (ll)zh * p.ash;
  const float* __restrict__ Bg = p.Bm + (ll)zb * p.bsb + (ll)zh * p.bsh;
  float*       __restrict__ C  = p.C  + (ll)zb * p.csb + (ll)zh * p.csh;
  const float* bias = p.bias ? p.bias + (ll)zb * p.biasb + (ll)zh * p.biash : nullptr;
  const float* res  = p.res  ? p.res  + (ll)zb * p.resb  + (ll)zh * p.resh  : nullptr;
  const float* mrow = p.mrow ? p.mrow + (ll)zb * p.mrowb : nullptr;
  const float* mcol = p.mcol ? p.mcol + (ll)zb * p.mcolb : nullptr;

  const int tid  = threadIdx.x;
  const int lane = tid & 31;
  const int wave = tid >> 5;      // 0..7
  const int wm   = wave & 3;      // 4 row tiles of 16  -> 64 rows
  const int wn   = wave >> 2;     // 2 col halves of 32 -> 64 cols
  const int m0   = blockIdx.x * TBM;
  const int n0   = blockIdx.y * TBN;
  const int hi   = lane >> 4;     // lane half
  const int ml   = lane & 15;

  v8f acc0 = {}, acc1 = {};

  for (int k0 = 0; k0 < p.K; k0 += TBK) {
    // speculative prefetch of next A tile (dropped silently if OOB)
    __builtin_prefetch(A + (ll)(m0 + (tid & 63)) * p.lda + k0 + TBK, 0, 3);

    // ---- stage A tile (64 x 32 f32) ----
#if HAVE_ASYNC_LDS
    // per-lane 16B async DMA copies, tracked with ASYNCcnt
#pragma unroll
    for (int it = 0; it < 2; ++it) {
      int q = tid + it * 256;          // 0..511 16B chunks (64 rows x 8)
      int r = q >> 3;
      int c = (q & 7) << 2;
      const float* gp = A + (ll)(m0 + r) * p.lda + (k0 + c);
      float* lp = &As32[r * ALD32 + c];
      __builtin_amdgcn_global_load_async_to_lds_b128(
          (as1_v4i)(unsigned long long)gp,
          (as3_v4i)(unsigned)(unsigned long long)lp,
          0, 0);
    }
#else
#pragma unroll
    for (int it = 0; it < 2; ++it) {
      int q = tid + it * 256;          // 0..511 float4 slots
      int r = q >> 3;
      int c = (q & 7) << 2;
      const float4 f = *(const float4*)(A + (ll)(m0 + r) * p.lda + (k0 + c));
      v4h h; h[0] = (_Float16)f.x; h[1] = (_Float16)f.y;
             h[2] = (_Float16)f.z; h[3] = (_Float16)f.w;
      *(v4h*)(&As[r * ALD + c]) = h;
    }
#endif
    // ---- stage B into Bs[n][k] f16 ----
    if (TRANSB) {
      // B global is [N][K] row-major: contiguous along k
#pragma unroll
      for (int it = 0; it < 2; ++it) {
        int q = tid + it * 256;        // 64 n x 8 k-quads
        int n = q >> 3;
        int c = (q & 7) << 2;
        const float4 f = *(const float4*)(Bg + (ll)(n0 + n) * p.ldb + (k0 + c));
        v4h h; h[0] = (_Float16)f.x; h[1] = (_Float16)f.y;
               h[2] = (_Float16)f.z; h[3] = (_Float16)f.w;
        *(v4h*)(&Bs[n * BLD + c]) = h;
      }
    } else {
      // B global is [K][N] row-major: read float4 along n, transpose into Bs
#pragma unroll
      for (int it = 0; it < 2; ++it) {
        int q = tid + it * 256;        // 32 k x 16 n-quads
        int k = q >> 4;
        int n = (q & 15) << 2;
        const float4 f = *(const float4*)(Bg + (ll)(k0 + k) * p.ldb + (n0 + n));
        Bs[(n + 0) * BLD + k] = (_Float16)f.x;
        Bs[(n + 1) * BLD + k] = (_Float16)f.y;
        Bs[(n + 2) * BLD + k] = (_Float16)f.z;
        Bs[(n + 3) * BLD + k] = (_Float16)f.w;
      }
    }
#if HAVE_ASYNC_LDS
    async_wait0();                     // drain this wave's async LDS writes
#endif
    __syncthreads();

    // ---- A fragment (ISA 7.12.2 16-bit A): lane row = wm*16+ml,
    // elems 0..7 -> K = hi*8..+7 ; elems 8..15 -> K = 16+hi*8..+7
    const int arow = wm * 16 + ml;
    v16h af;
#if HAVE_ASYNC_LDS
    {
      const float* a32 = &As32[arow * ALD32];
      const float4 fa0 = *(const float4*)(a32 + hi * 8);
      const float4 fa1 = *(const float4*)(a32 + hi * 8 + 4);
      const float4 fa2 = *(const float4*)(a32 + 16 + hi * 8);
      const float4 fa3 = *(const float4*)(a32 + 16 + hi * 8 + 4);
      af[0]  = (_Float16)fa0.x; af[1]  = (_Float16)fa0.y;
      af[2]  = (_Float16)fa0.z; af[3]  = (_Float16)fa0.w;
      af[4]  = (_Float16)fa1.x; af[5]  = (_Float16)fa1.y;
      af[6]  = (_Float16)fa1.z; af[7]  = (_Float16)fa1.w;
      af[8]  = (_Float16)fa2.x; af[9]  = (_Float16)fa2.y;
      af[10] = (_Float16)fa2.z; af[11] = (_Float16)fa2.w;
      af[12] = (_Float16)fa3.x; af[13] = (_Float16)fa3.y;
      af[14] = (_Float16)fa3.z; af[15] = (_Float16)fa3.w;
    }
#else
    {
      v8h a0 = *(const v8h*)(&As[arow * ALD + hi * 8]);
      v8h a1 = *(const v8h*)(&As[arow * ALD + 16 + hi * 8]);
      af = __builtin_shufflevector(a0, a1, 0,1,2,3,4,5,6,7,8,9,10,11,12,13,14,15);
    }
#endif

    // ---- B fragments: lane col = ml, elems 0..7 -> K = hi*8..+7 ; elems 8..15 -> K = 16+hi*8..+7 ----
    const int bn0 = wn * 32 + ml;
    v8h b0 = *(const v8h*)(&Bs[bn0 * BLD + hi * 8]);
    v8h b1 = *(const v8h*)(&Bs[bn0 * BLD + 16 + hi * 8]);
    v16h bf0 = __builtin_shufflevector(b0, b1, 0,1,2,3,4,5,6,7,8,9,10,11,12,13,14,15);
    const int bn1 = bn0 + 16;
    v8h b2 = *(const v8h*)(&Bs[bn1 * BLD + hi * 8]);
    v8h b3 = *(const v8h*)(&Bs[bn1 * BLD + 16 + hi * 8]);
    v16h bf1 = __builtin_shufflevector(b2, b3, 0,1,2,3,4,5,6,7,8,9,10,11,12,13,14,15);

    acc0 = __builtin_amdgcn_wmma_f32_16x16x32_f16(false, af, false, bf0,
                                                  (short)0, acc0, false, false);
    acc1 = __builtin_amdgcn_wmma_f32_16x16x32_f16(false, af, false, bf1,
                                                  (short)0, acc1, false, false);
    __syncthreads();
  }

  // ---- epilogue: C layout lane -> N = ml, VGPR r -> M = r + 8*hi ----
  const int mbase = m0 + wm * 16 + hi * 8;
  auto epi = [&](const v8f& acc, int n) {
    float colAdd = 0.0f;
    if (bias) colAdd += bias[n];
    if (mcol) colAdd += mcol[n];
    float rowAdd[8];
#pragma unroll
    for (int r = 0; r < 8; ++r) rowAdd[r] = 0.0f;
    if (mrow) {
#pragma unroll
      for (int r = 0; r < 8; ++r) rowAdd[r] += mrow[mbase + r];
    }
    if (res) {
#pragma unroll
      for (int r = 0; r < 8; ++r) rowAdd[r] += res[(ll)(mbase + r) * p.ldc + n];
    }
#pragma unroll
    for (int r = 0; r < 8; ++r) {
      int m = mbase + r;
      float v = acc[r] * p.scale + colAdd + rowAdd[r];
      if (p.causal && n > m) v = NEGINF;
      if (p.relu) v = fmaxf(v, 0.0f);
      C[(ll)m * p.ldc + n] = v;
    }
  };
  epi(acc0, n0 + wn * 32 + ml);
  epi(acc1, n0 + wn * 32 + 16 + ml);
}

// ---------------- embedding + positional encoding + pad mask ----------------
__launch_bounds__(256)
__global__ void embed_pe(const int* __restrict__ tok, const int* __restrict__ padi,
                         const float* __restrict__ emb, float* __restrict__ out,
                         float* __restrict__ mask) {
  int bs = blockIdx.x;              // b*S + s
  int b = bs >> 9, s = bs & 511;
  int valid = s < padi[b];
  if (threadIdx.x == 0) mask[bs] = valid ? 0.0f : NEGINF;
  int t = tok[bs];
#pragma unroll
  for (int it = 0; it < 2; ++it) {
    int d = threadIdx.x + it * 256;
    float e = emb[(ll)t * DMM + d];
    float expo = (float)(2 * (d >> 1)) * (1.0f / (float)DMM);
    float ang = (float)s * __expf(-expo * 9.210340371976184f);   // / 10000^expo
    float pe = (d & 1) ? __cosf(ang) : __sinf(ang);
    out[(ll)bs * DMM + d] = valid ? (e + pe) : 0.0f;
  }
}

// ---------------- attention softmax: one wave per 512-col row ----------------
__launch_bounds__(256)
__global__ void attn_softmax(float* __restrict__ sc) {
  int wave = threadIdx.x >> 5, lane = threadIdx.x & 31;
  ll row = (ll)blockIdx.x * 8 + wave;
  float* pr = sc + row * SS;
  float v[16];
  float mx = NEGINF;
#pragma unroll
  for (int i = 0; i < 16; ++i) { v[i] = pr[lane + i * 32]; mx = fmaxf(mx, v[i]); }
#pragma unroll
  for (int off = 16; off >= 1; off >>= 1) mx = fmaxf(mx, __shfl_xor(mx, off, 32));
  if (mx == NEGINF) mx = 0.0f;          // csoftmax: all -inf row -> max treated as 0
  float s = 0.0f;
#pragma unroll
  for (int i = 0; i < 16; ++i) { v[i] = __expf(v[i] - mx); s += v[i]; }
#pragma unroll
  for (int off = 16; off >= 1; off >>= 1) s += __shfl_xor(s, off, 32);
  float inv = 1.0f / (s + 1e-10f);
#pragma unroll
  for (int i = 0; i < 16; ++i) pr[lane + i * 32] = v[i] * inv;
}

// ---------------- layernorm: one 256-thread block per 512-col row ----------------
__launch_bounds__(256)
__global__ void layernorm_k(const float* __restrict__ in, float* __restrict__ out,
                            const float* __restrict__ g, const float* __restrict__ bb) {
  __shared__ float red[8];
  ll row = blockIdx.x;
  const float* pi = in + row * DMM;
  float* po = out + row * DMM;
  int tid = threadIdx.x, lane = tid & 31, wv = tid >> 5;
  float v0 = pi[tid], v1 = pi[tid + 256];
  float s = v0 + v1;
#pragma unroll
  for (int off = 16; off >= 1; off >>= 1) s += __shfl_xor(s, off, 32);
  if (lane == 0) red[wv] = s;
  __syncthreads();
  float tot = 0.0f;
#pragma unroll
  for (int i = 0; i < 8; ++i) tot += red[i];
  float mu = tot * (1.0f / (float)DMM);
  float d0 = v0 - mu, d1 = v1 - mu;
  float s2 = d0 * d0 + d1 * d1;
#pragma unroll
  for (int off = 16; off >= 1; off >>= 1) s2 += __shfl_xor(s2, off, 32);
  __syncthreads();
  if (lane == 0) red[wv] = s2;
  __syncthreads();
  float tot2 = 0.0f;
#pragma unroll
  for (int i = 0; i < 8; ++i) tot2 += red[i];
  float rs = rsqrtf(tot2 * (1.0f / (float)DMM) + 1e-5f);
  po[tid]       = g[tid]       * d0 * rs + bb[tid];
  po[tid + 256] = g[tid + 256] * d1 * rs + bb[tid + 256];
}

// ---------------- final masked vocab softmax (in-place on logits) ----------------
__launch_bounds__(256)
__global__ void vocab_softmax(float* __restrict__ logits, const int* __restrict__ yi) {
  __shared__ float redm[8];
  __shared__ float reds[8];
  int b = blockIdx.x >> 9, t = blockIdx.x & 511;
  float* pr = logits + (ll)blockIdx.x * VV;
  int pad = t >= yi[b];
  int lane = threadIdx.x & 31, wv = threadIdx.x >> 5;

  float mx = NEGINF;
  for (int c = threadIdx.x; c < VV; c += 256) {
    float lv = pr[c];
    float v = pad ? ((c == 0) ? lv : NEGINF) : ((c == 0) ? NEGINF : lv);
    mx = fmaxf(mx, v);
  }
#pragma unroll
  for (int off = 16; off >= 1; off >>= 1) mx = fmaxf(mx, __shfl_xor(mx, off, 32));
  if (lane == 0) redm[wv] = mx;
  __syncthreads();
  mx = redm[0];
#pragma unroll
  for (int i = 1; i < 8; ++i) mx = fmaxf(mx, redm[i]);
  if (mx == NEGINF) mx = 0.0f;

  float s = 0.0f;
  for (int c = threadIdx.x; c < VV; c += 256) {
    float lv = pr[c];
    float v = pad ? ((c == 0) ? lv : NEGINF) : ((c == 0) ? NEGINF : lv);
    s += __expf(v - mx);
  }
#pragma unroll
  for (int off = 16; off >= 1; off >>= 1) s += __shfl_xor(s, off, 32);
  if (lane == 0) reds[wv] = s;
  __syncthreads();
  s = 0.0f;
#pragma unroll
  for (int i = 0; i < 8; ++i) s += reds[i];
  float inv = 1.0f / (s + 1e-10f);

  for (int c = threadIdx.x; c < VV; c += 256) {
    float lv = pr[c];
    float v = pad ? ((c == 0) ? lv : NEGINF) : ((c == 0) ? NEGINF : lv);
    pr[c] = __expf(v - mx) * inv;
  }
}

// ---------------- host orchestration ----------------
struct MhaW { const float *wq,*bq,*wk,*bk,*wv,*bv,*wo,*bo; };
struct BlkW { MhaW m; const float *ln1g,*ln1b,*w1,*b1,*w2,*b2,*ln2g,*ln2b; };

extern "C" void kernel_launch(void* const* d_in, const int* in_sizes, int n_in,
                              void* d_out, int out_size, void* d_ws, size_t ws_size,
                              hipStream_t stream) {
  (void)in_sizes; (void)n_in; (void)out_size; (void)ws_size;

  const int*   xb  = (const int*)d_in[0];
  const int*   xpi = (const int*)d_in[1];
  const int*   ybt = (const int*)d_in[2];
  const int*   ypi = (const int*)d_in[3];
  const float* emb = (const float*)d_in[4];
  auto F = [&](int i) { return (const float*)d_in[i]; };
  auto mhaAt = [&](int base) -> MhaW {
    return { F(base+0),F(base+1),F(base+2),F(base+3),F(base+4),F(base+5),F(base+6),F(base+7) };
  };
  auto blkAt = [&](int base) -> BlkW {
    BlkW w; w.m = mhaAt(base);
    w.ln1g=F(base+8); w.ln1b=F(base+9); w.w1=F(base+10); w.b1=F(base+11);
    w.w2=F(base+12); w.b2=F(base+13); w.ln2g=F(base+14); w.ln2b=F(base+15);
    return w;
  };

  // workspace layout (floats)
  float* ws = (float*)d_ws;
  const ll BSD = (ll)BB * SS * DMM;      // 524288
  ll o = 0;
  float* X  = ws + o; o += BSD;
  float* Y  = ws + o; o += BSD;
  float* XM = ws + o; o += (ll)BB * SS;
  float* YM = ws + o; o += (ll)BB * SS;
  float* Q  = ws + o; o += BSD;
  float* Kb = ws + o; o += BSD;
  float* Vb = ws + o; o += BSD;
  float* SC = ws + o; o += (ll)BB * HH * SS * SS;
  float* CC = ws + o; o += BSD;
  float* T0 = ws + o; o += BSD;
  float* T1 = ws + o; o += BSD;
  float* T2 = ws + o; o += BSD;
  float* F1 = ws + o; o += (ll)BB * SS * FFF;

  auto gl = [&](const GemmP& p, int nz) {
    dim3 grid(p.M / TBM, p.N / TBN, nz);
    if (p.transB) gemm_wmma<1><<<grid, 256, 0, stream>>>(p);
    else          gemm_wmma<0><<<grid, 256, 0, stream>>>(p);
  };

  // embeddings + PE + pad masks
  embed_pe<<<BB * SS, 256, 0, stream>>>(xb,  xpi, emb, X, XM);
  embed_pe<<<BB * SS, 256, 0, stream>>>(ybt, ypi, emb, Y, YM);

  // full multi-head attention (projections, scores+mask, softmax, AV, out-proj+residual)
  auto run_mha = [&](const float* qin, const float* kvin, const MhaW& w,
                     const float* mr, const float* mc, int causal,
                     const float* resid, float* outBuf) {
    auto proj = [&](const float* inp, const float* wgt, const float* bias, float* out) {
      GemmP p{}; p.A = inp; p.asb = (ll)SS * DMM; p.lda = DMM;
      p.Bm = wgt; p.bsh = (ll)DMM * DKK; p.ldb = DKK;
      p.C = out; p.csb = (ll)HH * SS * DKK; p.csh = (ll)SS * DKK; p.ldc = DKK;
      p.bias = bias; p.biash = DKK;
      p.M = SS; p.N = DKK; p.K = DMM; p.hshift = 3; p.scale = 1.0f;
      gl(p, BB * HH);
    };
    proj(qin,  w.wq, w.bq, Q);
    proj(kvin, w.wk, w.bk, Kb);
    proj(kvin, w.wv, w.bv, Vb);
    { // scores = Q K^T / sqrt(dk) + rowmask + colmask (+causal)
      GemmP p{}; p.A = Q; p.asb = (ll)HH * SS * DKK; p.ash = (ll)SS * DKK; p.lda = DKK;
      p.Bm = Kb; p.bsb = (ll)HH * SS * DKK; p.bsh = (ll)SS * DKK; p.ldb = DKK; p.transB = 1;
      p.C = SC; p.csb = (ll)HH * SS * SS; p.csh = (ll)SS * SS; p.ldc = SS;
      p.mrow = mr; p.mrowb = SS; p.mcol = mc; p.mcolb = SS;
      p.M = SS; p.N = SS; p.K = DKK; p.hshift = 3; p.scale = 0.125f; p.causal = causal;
      gl(p, BB * HH);
    }
    attn_softmax<<<(BB * HH * SS) / 8, 256, 0, stream>>>(SC);
    { // O = P V  (written directly into head-concat layout)
      GemmP p{}; p.A = SC; p.asb = (ll)HH * SS * SS; p.ash = (ll)SS * SS; p.lda = SS;
      p.Bm = Vb; p.bsb = (ll)HH * SS * DKK; p.bsh = (ll)SS * DKK; p.ldb = DVV;
      p.C = CC; p.csb = (ll)SS * DMM; p.csh = DVV; p.ldc = DMM;
      p.M = SS; p.N = DVV; p.K = SS; p.hshift = 3; p.scale = 1.0f;
      gl(p, BB * HH);
    }
    { // out proj + bias + residual
      GemmP p{}; p.A = CC; p.asb = (ll)SS * DMM; p.lda = DMM;
      p.Bm = w.wo; p.ldb = DMM;
      p.C = outBuf; p.csb = (ll)SS * DMM; p.ldc = DMM;
      p.bias = w.bo;
      p.res = resid; p.resb = (ll)SS * DMM;
      p.M = SS; p.N = DMM; p.K = HH * DVV; p.hshift = 0; p.scale = 1.0f;
      gl(p, BB);
    }
  };

  auto run_ffn = [&](const float* nin, const float* w1, const float* b1,
                     const float* w2, const float* b2, float* outBuf) {
    { GemmP p{}; p.A = nin; p.asb = (ll)SS * DMM; p.lda = DMM;
      p.Bm = w1; p.ldb = FFF;
      p.C = F1; p.csb = (ll)SS * FFF; p.ldc = FFF;
      p.bias = b1; p.relu = 1;
      p.M = SS; p.N = FFF; p.K = DMM; p.hshift = 0; p.scale = 1.0f;
      gl(p, BB); }
    { GemmP p{}; p.A = F1; p.asb = (ll)SS * FFF; p.lda = FFF;
      p.Bm = w2; p.ldb = DMM;
      p.C = outBuf; p.csb = (ll)SS * DMM; p.ldc = DMM;
      p.bias = b2;
      p.res = nin; p.resb = (ll)SS * DMM;
      p.M = SS; p.N = DMM; p.K = FFF; p.hshift = 0; p.scale = 1.0f;
      gl(p, BB); }
  };

  // ---- encoder ----
  for (int i = 0; i < NBB; ++i) {
    BlkW w = blkAt(5 + 16 * i);
    run_mha(X, X, w.m, XM, XM, /*causal=*/0, /*res=*/X, T0);
    layernorm_k<<<BB * SS, 256, 0, stream>>>(T0, T1, w.ln1g, w.ln1b);
    run_ffn(T1, w.w1, w.b1, w.w2, w.b2, T0);
    layernorm_k<<<BB * SS, 256, 0, stream>>>(T0, X, w.ln2g, w.ln2b);
  }

  // ---- decoder ----
  for (int i = 0; i < NBB; ++i) {
    int base = 5 + 16 * NBB + 26 * i;
    MhaW self = mhaAt(base);
    const float* lng = F(base + 8);
    const float* lnb = F(base + 9);
    BlkW cr = blkAt(base + 10);

    // masked self-attention
    run_mha(Y, Y, self, YM, YM, /*causal=*/1, /*res=*/Y, T0);
    layernorm_k<<<BB * SS, 256, 0, stream>>>(T0, T1, lng, lnb);         // n
    // cross-attention block (q = n, k = v = encoder output X)
    run_mha(T1, X, cr.m, YM, XM, /*causal=*/0, /*res=*/T1, T0);
    layernorm_k<<<BB * SS, 256, 0, stream>>>(T0, T2, cr.ln1g, cr.ln1b); // n1
    run_ffn(T2, cr.w1, cr.b1, cr.w2, cr.b2, T0);
    layernorm_k<<<BB * SS, 256, 0, stream>>>(T0, Y, cr.ln2g, cr.ln2b);
  }

  // ---- tied-embedding logits (directly into d_out), then masked softmax ----
  { GemmP p{}; p.A = Y; p.asb = (ll)SS * DMM; p.lda = DMM;
    p.Bm = emb; p.ldb = DMM; p.transB = 1;
    p.C = (float*)d_out; p.csb = (ll)SS * VV; p.ldc = VV;
    p.M = SS; p.N = VV; p.K = DMM; p.hshift = 0; p.scale = 1.0f;
    gl(p, BB); }
  vocab_softmax<<<BB * SS, 256, 0, stream>>>((float*)d_out, ypi);
}
